// Net_69887707841036
// MI455X (gfx1250) — matrix operand order repaired and checked by reading, not compiled, not run on hardware
//
#include <hip/hip_runtime.h>
#include <math.h>
#include <stdint.h>

// ---------------------------------------------------------------------------
// Pegasus-XSUM style encoder/decoder generation for MI455X (gfx1250).
//
// * All dense GEMMs run through v_wmma_f32_16x16x32_f16 (f32 accumulate).
//   Each wave computes a 16x64 output strip (4 accumulators): the A operand
//   is fetched/converted once per 32-wide k-step and reused by 4 WMMAs.
//   The k-loop is software-pipelined one deep in registers: iteration k+1's
//   A and B tiles are loaded before iteration k's WMMAs issue, so the WMMAs
//   overlap the in-flight global_load_b128s instead of waiting to zero.
// * Weights (and the embedding matrix E) are pre-converted once per call to
//   f16 in B^T ([N,K]) layout, so every WMMA B-operand fetch is one aligned
//   32-byte contiguous load (2x global_load_b128) and decode-dominant HBM
//   traffic is halved: logits GEMM streams E-f16 (197MB) instead of fp32
//   (394MB) per step -> decode floor ~80*197MB/23.3TB/s ~ 0.68ms.
// * No __builtin_prefetch: it forces generic addressing (flat_load_* +
//   loadcnt/dscnt joint waits) which serializes the B stream.
// * Out-of-range tile rows/cols are index-clamped (never zero-filled):
//   D[m,n] depends only on A row m / B col n, and unsaved outputs are
//   discarded, so the inner loop has no cndmask/exec-mask operations.
// * Attention is flash-style online softmax (no [B,H,S,S] materialization).
//
// Workspace requirement: ~450 MB (fp32 activations + f16 weight shadows).
// ---------------------------------------------------------------------------

namespace cfg {
constexpr int V = 96103, D = 1024, H = 16, DH = 64, F = 4096, L = 2;
constexpr int S = 512, T = 80, B = 4, NS = 5, BN = B * NS, TOPK = 100;
constexpr int EOS = 1, MINLEN = 7;
constexpr float EMB_SCALE = 32.0f;       // sqrt(1024)
constexpr float INV_DH = 0.125f;         // 1/sqrt(64)
}
using namespace cfg;

typedef __attribute__((ext_vector_type(16))) _Float16 v16h;
typedef __attribute__((ext_vector_type(8)))  float    v8f;

__device__ __forceinline__ float wave_sum32(float v) {
#pragma unroll
  for (int off = 16; off > 0; off >>= 1) v += __shfl_xor(v, off, 32);
  return v;
}

__device__ __forceinline__ float gelu_f(float x) {  // jax.nn.gelu (tanh approx)
  const float c = 0.7978845608028654f;
  float x3 = x * x * x;
  return 0.5f * x * (1.0f + tanhf(c * (x + 0.044715f * x3)));
}

__device__ __forceinline__ float rand01(uint32_t t, uint32_t bn) {
  uint32_t x = 1234u ^ (t * 0x9E3779B9u) ^ (bn * 0x85EBCA77u);
  x ^= x >> 16; x *= 0x7feb352du; x ^= x >> 15; x *= 0x846ca68bu; x ^= x >> 16;
  return (float)(x >> 8) * (1.0f / 16777216.0f);
}

// Load + convert the A 16x32 f16 operand (elems 0..15 = K +half*16+j, contiguous).
__device__ __forceinline__ v16h load_a_f16(const float* __restrict__ ap) {
  v16h a;
#pragma unroll
  for (int j = 0; j < 8; ++j) {
    a[j]     = (_Float16)ap[j];
    a[j + 8] = (_Float16)ap[j + 8];
  }
  return a;
}

// ---------------------------------------------------------------------------
// Weight pre-conversion: W[K,N] fp32 -> Wt[N,K] f16 (B^T layout for WMMA).
// ---------------------------------------------------------------------------
__global__ void transpose_to_f16_kernel(const float* __restrict__ W,
                                        _Float16* __restrict__ Wt,
                                        int K, int N) {
  const int k = blockIdx.x;
  const int n = blockIdx.y * blockDim.x + threadIdx.x;
  if (n < N) Wt[(size_t)n * K + k] = (_Float16)W[(size_t)k * N + n];
}

// Straight fp32 -> f16 copy (E is already [V,D] = [N,K] for h @ E^T).
__global__ void cvt_f16_kernel(const float* __restrict__ src,
                               _Float16* __restrict__ dst, size_t n) {
  size_t i = (size_t)blockIdx.x * blockDim.x + threadIdx.x;
  if (i < n) dst[i] = (_Float16)src[i];
}

// ---------------------------------------------------------------------------
// WMMA GEMM: C[M,N] = (act)(A[M,K] @ B) (+ Resid).
// One wave per 16x64 strip; software-pipelined one iteration deep.
// K must be a multiple of 32 (D=1024, F=4096).
// ---------------------------------------------------------------------------
__global__ void wmma_gemm_kernel(const float* __restrict__ A,
                                 const _Float16* __restrict__ Bt,
                                 const float* __restrict__ Resid,
                                 float* __restrict__ C,
                                 int M, int N, int K, int act) {
  const int lane = threadIdx.x;            // wave32
  const int half = lane >> 4;
  const int lidx = lane & 15;
  const int tileN = blockIdx.x * 64;
  const int tileM = blockIdx.y * 16;
  // Clamp indices for safe addressing; garbage rows/cols only affect
  // D elements that are never stored (see header comment).
  const int rowC = min(tileM + lidx, M - 1);
  const float* ap = A + (size_t)rowC * K + half * 16;   // 32B aligned

  const _Float16* bp0 = Bt + (size_t)min(tileN +      lidx, N - 1) * K + half * 16;
  const _Float16* bp1 = Bt + (size_t)min(tileN + 16 + lidx, N - 1) * K + half * 16;
  const _Float16* bp2 = Bt + (size_t)min(tileN + 32 + lidx, N - 1) * K + half * 16;
  const _Float16* bp3 = Bt + (size_t)min(tileN + 48 + lidx, N - 1) * K + half * 16;

  v8f acc0 = {0.f,0.f,0.f,0.f,0.f,0.f,0.f,0.f};
  v8f acc1 = acc0, acc2 = acc0, acc3 = acc0;

  // ---- prologue: generation 0 in registers ----
  v16h a  = load_a_f16(ap);
  v16h b0 = *(const v16h*)bp0;             // 2x global_load_b128 each
  v16h b1 = *(const v16h*)bp1;
  v16h b2 = *(const v16h*)bp2;
  v16h b3 = *(const v16h*)bp3;

  for (int kk = 32; kk < K; kk += 32) {
    ap += 32; bp0 += 32; bp1 += 32; bp2 += 32; bp3 += 32;
    // Issue next generation's loads before consuming the current one.
    v16h aN  = load_a_f16(ap);
    v16h c0 = *(const v16h*)bp0;
    v16h c1 = *(const v16h*)bp1;
    v16h c2 = *(const v16h*)bp2;
    v16h c3 = *(const v16h*)bp3;
    acc0 = __builtin_amdgcn_wmma_f32_16x16x32_f16(false, a, false, b0,
                                                  (short)0, acc0, false, false);
    acc1 = __builtin_amdgcn_wmma_f32_16x16x32_f16(false, a, false, b1,
                                                  (short)0, acc1, false, false);
    acc2 = __builtin_amdgcn_wmma_f32_16x16x32_f16(false, a, false, b2,
                                                  (short)0, acc2, false, false);
    acc3 = __builtin_amdgcn_wmma_f32_16x16x32_f16(false, a, false, b3,
                                                  (short)0, acc3, false, false);
    a = aN; b0 = c0; b1 = c1; b2 = c2; b3 = c3;
  }
  // ---- epilogue of the pipeline ----
  acc0 = __builtin_amdgcn_wmma_f32_16x16x32_f16(false, a, false, b0,
                                                (short)0, acc0, false, false);
  acc1 = __builtin_amdgcn_wmma_f32_16x16x32_f16(false, a, false, b1,
                                                (short)0, acc1, false, false);
  acc2 = __builtin_amdgcn_wmma_f32_16x16x32_f16(false, a, false, b2,
                                                (short)0, acc2, false, false);
  acc3 = __builtin_amdgcn_wmma_f32_16x16x32_f16(false, a, false, b3,
                                                (short)0, acc3, false, false);

  // C/D layout: VGPR r -> M = r + 8*half, lane%16 -> N.
  const v8f accs[4] = {acc0, acc1, acc2, acc3};
#pragma unroll
  for (int g = 0; g < 4; ++g) {
    const int col = tileN + g * 16 + lidx;
    if (col < N) {
#pragma unroll
      for (int r = 0; r < 8; ++r) {
        int m = tileM + half * 8 + r;
        if (m < M) {
          float v = accs[g][r];
          if (act == 1) v = gelu_f(v);
          if (Resid) v += Resid[(size_t)m * N + col];
          C[(size_t)m * N + col] = v;
        }
      }
    }
  }
}

// ---------------------------------------------------------------------------
// LayerNorm (no affine): one wave per row of D=1024.
// ---------------------------------------------------------------------------
__global__ void ln_kernel(const float* __restrict__ x, float* __restrict__ y) {
  const int r = blockIdx.x;
  const int lane = threadIdx.x;
  const float* xr = x + (size_t)r * D;
  float s = 0.f, s2 = 0.f;
  for (int j = lane; j < D; j += 32) { float v = xr[j]; s += v; s2 += v * v; }
  s = wave_sum32(s); s2 = wave_sum32(s2);
  const float mean = s * (1.0f / D);
  const float var = s2 * (1.0f / D) - mean * mean;
  const float rstd = rsqrtf(var + 1e-5f);
  float* yr = y + (size_t)r * D;
  for (int j = lane; j < D; j += 32) yr[j] = (xr[j] - mean) * rstd;
}

// ---------------------------------------------------------------------------
// Embedding kernels
// ---------------------------------------------------------------------------
__global__ void enc_embed_kernel(const int* __restrict__ ids,
                                 const float* __restrict__ E,
                                 const float* __restrict__ pos_e,
                                 float* __restrict__ x) {
  const int bs = blockIdx.x;             // 0..B*S-1
  const int s = bs % S;
  const int id = ids[bs];
  for (int j = threadIdx.x; j < D; j += blockDim.x)
    x[(size_t)bs * D + j] = E[(size_t)id * D + j] * EMB_SCALE + pos_e[(size_t)s * D + j];
}

__global__ void dec_embed_kernel(const int* __restrict__ tok_prev,
                                 const float* __restrict__ E,
                                 const float* __restrict__ pos_d,
                                 float* __restrict__ x, int t) {
  const int bn = blockIdx.x;
  const int id = tok_prev[bn];
  for (int j = threadIdx.x; j < D; j += blockDim.x)
    x[(size_t)bn * D + j] = E[(size_t)id * D + j] * EMB_SCALE + pos_d[(size_t)t * D + j];
}

// ---------------------------------------------------------------------------
// Encoder flash attention: one wave per (q, h, b); online softmax over S keys.
// qkv layout: [B*S, 3D], q | k | v slices; head h = cols h*DH..h*DH+63.
// ---------------------------------------------------------------------------
__global__ void enc_attn_kernel(const float* __restrict__ qkv,
                                const int* __restrict__ mask,
                                float* __restrict__ aout) {
  const int qpos = blockIdx.x, h = blockIdx.y, b = blockIdx.z;
  const int lane = threadIdx.x;
  const float* q = qkv + ((size_t)b * S + qpos) * (3 * D) + h * DH;
  const float q0 = q[2 * lane], q1 = q[2 * lane + 1];
  float m = -1e30f, l = 0.f, a0 = 0.f, a1 = 0.f;
  for (int k = 0; k < S; ++k) {
    const float* kp = qkv + ((size_t)b * S + k) * (3 * D) + D + h * DH;
    float sc = q0 * kp[2 * lane] + q1 * kp[2 * lane + 1];
    sc = wave_sum32(sc) * INV_DH + (mask[b * S + k] ? 0.f : -1e9f);
    const float mn = fmaxf(m, sc);
    const float corr = expf(m - mn);
    const float w = expf(sc - mn);
    const float* vp = kp + D;
    a0 = a0 * corr + w * vp[2 * lane];
    a1 = a1 * corr + w * vp[2 * lane + 1];
    l = l * corr + w;
    m = mn;
  }
  float* o = aout + ((size_t)b * S + qpos) * D + h * DH;
  o[2 * lane] = a0 / l;
  o[2 * lane + 1] = a1 / l;
}

// ---------------------------------------------------------------------------
// Decoder self-attention over cached K/V (positions 0..t).
// ---------------------------------------------------------------------------
__global__ void dec_self_attn_kernel(const float* __restrict__ qkv,
                                     const float* __restrict__ ck,
                                     const float* __restrict__ cv,
                                     float* __restrict__ aout, int t) {
  const int h = blockIdx.x, bn = blockIdx.y, lane = threadIdx.x;
  const float* q = qkv + (size_t)bn * (3 * D) + h * DH;
  const float q0 = q[2 * lane], q1 = q[2 * lane + 1];
  float m = -1e30f, l = 0.f, a0 = 0.f, a1 = 0.f;
  for (int tt = 0; tt <= t; ++tt) {
    const float* kp = ck + ((size_t)bn * T + tt) * D + h * DH;
    float sc = q0 * kp[2 * lane] + q1 * kp[2 * lane + 1];
    sc = wave_sum32(sc) * INV_DH;
    const float mn = fmaxf(m, sc);
    const float corr = expf(m - mn);
    const float w = expf(sc - mn);
    const float* vp = cv + ((size_t)bn * T + tt) * D + h * DH;
    a0 = a0 * corr + w * vp[2 * lane];
    a1 = a1 * corr + w * vp[2 * lane + 1];
    l = l * corr + w;
    m = mn;
  }
  float* o = aout + (size_t)bn * D + h * DH;
  o[2 * lane] = a0 / l;
  o[2 * lane + 1] = a1 / l;
}

// ---------------------------------------------------------------------------
// Decoder cross-attention over precomputed encoder K/V ([B,S,D] per layer).
// ---------------------------------------------------------------------------
__global__ void cross_attn_kernel(const float* __restrict__ q2,
                                  const float* __restrict__ cK,
                                  const float* __restrict__ cV,
                                  const int* __restrict__ mask,
                                  float* __restrict__ aout) {
  const int h = blockIdx.x, bn = blockIdx.y, lane = threadIdx.x;
  const int b = bn / NS;
  const float* q = q2 + (size_t)bn * D + h * DH;
  const float q0 = q[2 * lane], q1 = q[2 * lane + 1];
  float m = -1e30f, l = 0.f, a0 = 0.f, a1 = 0.f;
  for (int s = 0; s < S; ++s) {
    const float* kp = cK + ((size_t)b * S + s) * D + h * DH;
    float sc = q0 * kp[2 * lane] + q1 * kp[2 * lane + 1];
    sc = wave_sum32(sc) * INV_DH + (mask[b * S + s] ? 0.f : -1e9f);
    const float mn = fmaxf(m, sc);
    const float corr = expf(m - mn);
    const float w = expf(sc - mn);
    const float* vp = cV + ((size_t)b * S + s) * D + h * DH;
    a0 = a0 * corr + w * vp[2 * lane];
    a1 = a1 * corr + w * vp[2 * lane + 1];
    l = l * corr + w;
    m = mn;
  }
  float* o = aout + (size_t)bn * D + h * DH;
  o[2 * lane] = a0 / l;
  o[2 * lane + 1] = a1 / l;
}

// ---------------------------------------------------------------------------
// Write current step's k/v into the decode cache.
// ---------------------------------------------------------------------------
__global__ void cache_write_kernel(const float* __restrict__ qkv,
                                   float* __restrict__ ck,
                                   float* __restrict__ cv, int t) {
  const int bn = blockIdx.x;
  for (int j = threadIdx.x; j < D; j += blockDim.x) {
    ck[((size_t)bn * T + t) * D + j] = qkv[(size_t)bn * (3 * D) + D + j];
    cv[((size_t)bn * T + t) * D + j] = qkv[(size_t)bn * (3 * D) + 2 * D + j];
  }
}

__global__ void init_tok_kernel(int* tok) {
  if ((int)threadIdx.x < BN) tok[threadIdx.x] = 0;
}

// ---------------------------------------------------------------------------
// Top-k(100) + categorical sampling; destructive on the logits row.
// ---------------------------------------------------------------------------
__global__ void topk_sample_kernel(float* __restrict__ logits,
                                   int* __restrict__ tok_prev,
                                   int* __restrict__ out_toks, int t) {
  const int bn = blockIdx.x;
  const int tid = threadIdx.x;
  float* lr = logits + (size_t)bn * V;
  __shared__ float s_val[256];
  __shared__ int s_idx[256];
  __shared__ float topv[TOPK];
  __shared__ int topi[TOPK];

  if (tid == 0 && t < MINLEN - 1) lr[EOS] = -1e9f;   // min_length: block EOS
  __syncthreads();

  for (int it = 0; it < TOPK; ++it) {
    float best = -3.0e38f; int bidx = 0;
    for (int v = tid; v < V; v += 256) {
      float x = lr[v];
      if (x > best) { best = x; bidx = v; }
    }
    s_val[tid] = best; s_idx[tid] = bidx;
    __syncthreads();
    for (int off = 128; off > 0; off >>= 1) {
      if (tid < off && s_val[tid + off] > s_val[tid]) {
        s_val[tid] = s_val[tid + off]; s_idx[tid] = s_idx[tid + off];
      }
      __syncthreads();
    }
    if (tid == 0) {
      topv[it] = s_val[0]; topi[it] = s_idx[0];
      lr[s_idx[0]] = -3.0e38f;                        // exclude from next pass
    }
    __syncthreads();
  }

  if (tid == 0) {
    const float mx = topv[0];                         // sorted descending
    float sum = 0.f;
    for (int i = 0; i < TOPK; ++i) sum += expf(topv[i] - mx);
    float u = rand01((uint32_t)t, (uint32_t)bn) * sum;
    int sel = TOPK - 1;
    float c = 0.f;
    for (int i = 0; i < TOPK; ++i) {
      c += expf(topv[i] - mx);
      if (u <= c) { sel = i; break; }
    }
    const int tok = topi[sel];
    tok_prev[bn] = tok;
    out_toks[(size_t)bn * T + t] = tok;               // [BN,T] == reshape(B,NS,T)
  }
}

// ---------------------------------------------------------------------------
// Host-side launch helpers
// ---------------------------------------------------------------------------
static inline void launch_gemm(const float* A, const _Float16* Bt, const float* R,
                               float* C, int M, int N, int K, int act,
                               hipStream_t s) {
  dim3 g((N + 63) / 64, (M + 15) / 16);
  wmma_gemm_kernel<<<g, 32, 0, s>>>(A, Bt, R, C, M, N, K, act);
}

static inline void launch_transpose(const float* W, _Float16* Wt, int K, int N,
                                    hipStream_t s) {
  dim3 g(K, (N + 255) / 256);
  transpose_to_f16_kernel<<<g, 256, 0, s>>>(W, Wt, K, N);
}

extern "C" void kernel_launch(void* const* d_in, const int* in_sizes, int n_in,
                              void* d_out, int out_size, void* d_ws, size_t ws_size,
                              hipStream_t stream) {
  (void)in_sizes; (void)n_in; (void)out_size; (void)ws_size;
  const int*   ids          = (const int*)d_in[0];
  const int*   mask         = (const int*)d_in[1];
  const float* E            = (const float*)d_in[2];
  const float* pos_e        = (const float*)d_in[3];
  const float* pos_d        = (const float*)d_in[4];
  const float* enc_attn_in  = (const float*)d_in[5];
  const float* enc_attn_out = (const float*)d_in[6];
  const float* enc_ffn1     = (const float*)d_in[7];
  const float* enc_ffn2     = (const float*)d_in[8];
  const float* dec_sa_in    = (const float*)d_in[9];
  const float* dec_sa_out   = (const float*)d_in[10];
  const float* dec_ca_q     = (const float*)d_in[11];
  const float* dec_ca_kv    = (const float*)d_in[12];
  const float* dec_ca_out   = (const float*)d_in[13];
  const float* dec_ffn1     = (const float*)d_in[14];
  const float* dec_ffn2     = (const float*)d_in[15];
  int* out_toks = (int*)d_out;

  // ---- workspace carve-up: fp32 activations, then int, then f16 shadows ----
  float* ws = (float*)d_ws;
  size_t o = 0;
  float* X      = ws + o; o += (size_t)B * S * D;       // encoder residual stream
  float* Hb     = ws + o; o += (size_t)B * S * D;       // ln output
  float* Tmp    = ws + o; o += (size_t)B * S * F;       // qkv / ffn1 scratch
  float* Abuf   = ws + o; o += (size_t)B * S * D;       // attention output
  float* cK     = ws + o; o += (size_t)L * B * S * D;   // cross-attn K
  float* cV     = ws + o; o += (size_t)L * B * S * D;   // cross-attn V
  float* ckc    = ws + o; o += (size_t)L * BN * T * D;  // decode K cache
  float* cvc    = ws + o; o += (size_t)L * BN * T * D;  // decode V cache
  float* xdec   = ws + o; o += (size_t)BN * D;
  float* hdec   = ws + o; o += (size_t)BN * D;
  float* qkvdec = ws + o; o += (size_t)BN * 3 * D;
  float* adec   = ws + o; o += (size_t)BN * D;
  float* q2dec  = ws + o; o += (size_t)BN * D;
  float* f1dec  = ws + o; o += (size_t)BN * F;
  float* logitsB= ws + o; o += (size_t)BN * V;
  int* tok_prev = (int*)(ws + o); o += 32;              // BN ints (padded)

  _Float16* h = (_Float16*)(ws + o);
  size_t q = 0;
  _Float16* Ef16      = h + q; q += (size_t)V * D;          // [V,D] == [N,K]
  _Float16* eAttnInT  = h + q; q += (size_t)L * 3 * D * D;  // [3D,D] per layer
  _Float16* eAttnOutT = h + q; q += (size_t)L * D * D;
  _Float16* eFfn1T    = h + q; q += (size_t)L * F * D;
  _Float16* eFfn2T    = h + q; q += (size_t)L * D * F;
  _Float16* dSaInT    = h + q; q += (size_t)L * 3 * D * D;
  _Float16* dSaOutT   = h + q; q += (size_t)L * D * D;
  _Float16* dCaQT     = h + q; q += (size_t)L * D * D;
  _Float16* dCaKvT    = h + q; q += (size_t)L * 2 * D * D;  // K rows 0..D-1, V rows D..2D-1
  _Float16* dCaOutT   = h + q; q += (size_t)L * D * D;
  _Float16* dFfn1T    = h + q; q += (size_t)L * F * D;
  _Float16* dFfn2T    = h + q; q += (size_t)L * D * F;

  const int M_enc = B * S;

  // ================= weight shadow conversion (one pass) =================
  {
    size_t nE = (size_t)V * D;
    cvt_f16_kernel<<<(int)((nE + 255) / 256), 256, 0, stream>>>(E, Ef16, nE);
    for (int l = 0; l < L; ++l) {
      launch_transpose(enc_attn_in  + (size_t)l * D * 3 * D, eAttnInT  + (size_t)l * 3 * D * D, D, 3 * D, stream);
      launch_transpose(enc_attn_out + (size_t)l * D * D,     eAttnOutT + (size_t)l * D * D,     D, D,     stream);
      launch_transpose(enc_ffn1     + (size_t)l * D * F,     eFfn1T    + (size_t)l * F * D,     D, F,     stream);
      launch_transpose(enc_ffn2     + (size_t)l * F * D,     eFfn2T    + (size_t)l * D * F,     F, D,     stream);
      launch_transpose(dec_sa_in    + (size_t)l * D * 3 * D, dSaInT    + (size_t)l * 3 * D * D, D, 3 * D, stream);
      launch_transpose(dec_sa_out   + (size_t)l * D * D,     dSaOutT   + (size_t)l * D * D,     D, D,     stream);
      launch_transpose(dec_ca_q     + (size_t)l * D * D,     dCaQT     + (size_t)l * D * D,     D, D,     stream);
      launch_transpose(dec_ca_kv    + (size_t)l * D * 2 * D, dCaKvT    + (size_t)l * 2 * D * D, D, 2 * D, stream);
      launch_transpose(dec_ca_out   + (size_t)l * D * D,     dCaOutT   + (size_t)l * D * D,     D, D,     stream);
      launch_transpose(dec_ffn1     + (size_t)l * D * F,     dFfn1T    + (size_t)l * F * D,     D, F,     stream);
      launch_transpose(dec_ffn2     + (size_t)l * F * D,     dFfn2T    + (size_t)l * D * F,     F, D,     stream);
    }
  }

  // ================= encoder =================
  enc_embed_kernel<<<M_enc, 256, 0, stream>>>(ids, E, pos_e, X);
  for (int l = 0; l < L; ++l) {
    ln_kernel<<<M_enc, 32, 0, stream>>>(X, Hb);
    launch_gemm(Hb, eAttnInT + (size_t)l * 3 * D * D, nullptr, Tmp,
                M_enc, 3 * D, D, 0, stream);
    enc_attn_kernel<<<dim3(S, H, B), 32, 0, stream>>>(Tmp, mask, Abuf);
    launch_gemm(Abuf, eAttnOutT + (size_t)l * D * D, X, X, M_enc, D, D, 0, stream);
    ln_kernel<<<M_enc, 32, 0, stream>>>(X, Hb);
    launch_gemm(Hb, eFfn1T + (size_t)l * F * D, nullptr, Tmp,
                M_enc, F, D, /*gelu*/1, stream);
    launch_gemm(Tmp, eFfn2T + (size_t)l * D * F, X, X, M_enc, D, F, 0, stream);
  }
  ln_kernel<<<M_enc, 32, 0, stream>>>(X, Hb);   // enc = ln(x)

  // cross-attention K/V per decoder layer: enc @ dec_ca_kv[l][:, :D | D:]
  for (int l = 0; l < L; ++l) {
    launch_gemm(Hb, dCaKvT + (size_t)l * 2 * D * D,                 nullptr,
                cK + (size_t)l * B * S * D, M_enc, D, D, 0, stream);
    launch_gemm(Hb, dCaKvT + (size_t)l * 2 * D * D + (size_t)D * D, nullptr,
                cV + (size_t)l * B * S * D, M_enc, D, D, 0, stream);
  }

  // ================= decode loop =================
  init_tok_kernel<<<1, 32, 0, stream>>>(tok_prev);
  for (int t = 0; t < T; ++t) {
    dec_embed_kernel<<<BN, 256, 0, stream>>>(tok_prev, E, pos_d, xdec, t);
    for (int l = 0; l < L; ++l) {
      float* ckl = ckc + (size_t)l * BN * T * D;
      float* cvl = cvc + (size_t)l * BN * T * D;
      // --- self-attention ---
      ln_kernel<<<BN, 32, 0, stream>>>(xdec, hdec);
      launch_gemm(hdec, dSaInT + (size_t)l * 3 * D * D, nullptr, qkvdec,
                  BN, 3 * D, D, 0, stream);
      cache_write_kernel<<<BN, 256, 0, stream>>>(qkvdec, ckl, cvl, t);
      dec_self_attn_kernel<<<dim3(H, BN), 32, 0, stream>>>(qkvdec, ckl, cvl, adec, t);
      launch_gemm(adec, dSaOutT + (size_t)l * D * D, xdec, xdec, BN, D, D, 0, stream);
      // --- cross-attention ---
      ln_kernel<<<BN, 32, 0, stream>>>(xdec, hdec);
      launch_gemm(hdec, dCaQT + (size_t)l * D * D, nullptr, q2dec, BN, D, D, 0, stream);
      cross_attn_kernel<<<dim3(H, BN), 32, 0, stream>>>(
          q2dec, cK + (size_t)l * B * S * D, cV + (size_t)l * B * S * D, mask, adec);
      launch_gemm(adec, dCaOutT + (size_t)l * D * D, xdec, xdec, BN, D, D, 0, stream);
      // --- FFN ---
      ln_kernel<<<BN, 32, 0, stream>>>(xdec, hdec);
      launch_gemm(hdec, dFfn1T + (size_t)l * F * D, nullptr, f1dec,
                  BN, F, D, /*gelu*/1, stream);
      launch_gemm(f1dec, dFfn2T + (size_t)l * D * F, xdec, xdec, BN, D, F, 0, stream);
    }
    // --- logits = ln(x) @ E^T, then top-k sample ---
    ln_kernel<<<BN, 32, 0, stream>>>(xdec, hdec);
    launch_gemm(hdec, Ef16, nullptr, logitsB, BN, V, D, 0, stream);
    topk_sample_kernel<<<BN, 256, 0, stream>>>(logitsB, tok_prev, out_toks, t);
  }
}
